// SequenceGenerator_16303695855874
// MI455X (gfx1250) — matrix operand order repaired and checked by reading, not compiled, not run on hardware
//
#include <hip/hip_runtime.h>

#define C 2048
#define BUF 16
#define ZPAD 8                      // leading zero rows: max dilation
#define NB 16                       // persistent grid: 16 blocks
#define TPB 256                     // 8 waves/block
#define NWAVES (NB * TPB / 32)      // 128 waves == 128 M-tiles
#define MAT_ELEMS ((size_t)C * C)
#define ZROWS (ZPAD + BUF)          // padded activation buffer rows

typedef __attribute__((ext_vector_type(8)))  float          v8f;
typedef __attribute__((ext_vector_type(16))) __bf16         v16bf;
typedef __attribute__((ext_vector_type(8)))  unsigned short u16x8;
typedef __attribute__((ext_vector_type(16))) unsigned short u16x16;

__device__ __forceinline__ unsigned short f2bf(float f) {
    unsigned u = __builtin_bit_cast(unsigned, f);
    unsigned r = ((u >> 16) & 1u) + 0x7FFFu;           // round-to-nearest-even
    return (unsigned short)((u + r) >> 16);
}
__device__ __forceinline__ float bf2f(unsigned short h) {
    return __builtin_bit_cast(float, ((unsigned)h) << 16);
}
__device__ __forceinline__ v16bf mk16(u16x8 lo, u16x8 hi) {
    u16x16 u;
#pragma unroll
    for (int i = 0; i < 8; ++i) { u[i] = lo[i]; u[i + 8] = hi[i]; }
    return __builtin_bit_cast(v16bf, u);
}
__device__ __forceinline__ v16bf as16(u16x16 u) { return __builtin_bit_cast(v16bf, u); }

// ---------------- grid-wide software barrier (persistent kernel) ------------
__device__ __forceinline__ void gsync(unsigned* bar) {
    __threadfence();
    __syncthreads();
    if (threadIdx.x == 0) {
        volatile unsigned* cnt = bar;
        volatile unsigned* gen = bar + 1;
        unsigned g = *gen;
        if (atomicAdd(bar, 1u) == gridDim.x - 1) {
            *cnt = 0;
            __threadfence();
            atomicAdd(bar + 1, 1u);
        } else {
            while (*gen == g) { __builtin_amdgcn_s_sleep(1); }
        }
        __threadfence();
    }
    __syncthreads();
}

// ---------------- one dilated conv layer as a 2048x16x2048 GEMM -------------
// zin/zout point at the BODY (row 0) of a buffer that has ZPAD zeroed rows
// physically in front, so the shifted tap (row n-d, d<=8) is always loadable.
// y = W0*shift(z,d) + W1*z + b + z, then leaky(0.2)
__device__ __forceinline__ void conv_layer(const unsigned short* __restrict__ W0,
                                           const unsigned short* __restrict__ W1,
                                           const float* __restrict__ bias,
                                           const unsigned short* __restrict__ zin,
                                           unsigned short* __restrict__ zout,
                                           int d, int wave, int lane) {
    for (int tile = wave; tile < 128; tile += NWAVES) {
        const int m0   = tile * 16;
        const int n    = lane & 15;    // B column == A row-in-tile
        const int half = lane >> 4;
        const unsigned short* a0row = W0 + (size_t)(m0 + n) * C;
        const unsigned short* a1row = W1 + (size_t)(m0 + n) * C;
        const unsigned short* b0row = zin + (ptrdiff_t)(n - d) * C;  // pad-backed
        const unsigned short* b1row = zin + (size_t)n * C;
        v8f acc = {};
#pragma unroll 4
        for (int kb = 0; kb < C; kb += 32) {
            __builtin_prefetch(a0row + kb + 128, 0, 3);
            __builtin_prefetch(a1row + kb + 128, 0, 3);
            v16bf a0 = mk16(*(const u16x8*)(a0row + kb + half * 8),
                            *(const u16x8*)(a0row + kb + 16 + half * 8));
            v16bf a1 = mk16(*(const u16x8*)(a1row + kb + half * 8),
                            *(const u16x8*)(a1row + kb + 16 + half * 8));
            v16bf b0u = mk16(*(const u16x8*)(b0row + kb + half * 8),
                             *(const u16x8*)(b0row + kb + 16 + half * 8));
            v16bf b1u = mk16(*(const u16x8*)(b1row + kb + half * 8),
                             *(const u16x8*)(b1row + kb + 16 + half * 8));
            acc = __builtin_amdgcn_wmma_f32_16x16x32_bf16(false, a0, false, b0u,
                                                          (short)0, acc, false, false);
            acc = __builtin_amdgcn_wmma_f32_16x16x32_bf16(false, a1, false, b1u,
                                                          (short)0, acc, false, false);
        }
#pragma unroll
        for (int r = 0; r < 8; ++r) {
            int m = m0 + half * 8 + r;
            float v = acc[r] + bias[m] + bf2f(zin[(size_t)n * C + m]);
            v = v > 0.f ? v : 0.2f * v;
            zout[(size_t)n * C + m] = f2bf(v);
        }
    }
}

// ---------------- matvec (res / out layers) via WMMA ------------------------
__device__ __forceinline__ void matvec(const unsigned short* __restrict__ W,
                                       const float* __restrict__ bias,
                                       const unsigned short* __restrict__ vin_b,
                                       const float* __restrict__ res_f,    // may be null
                                       unsigned short* __restrict__ out_b, // may be null
                                       float* __restrict__ out_f,          // may be null
                                       int leaky, int wave, int lane) {
    for (int tile = wave; tile < 128; tile += NWAVES) {
        const int m0   = tile * 16;
        const int n    = lane & 15;
        const int half = lane >> 4;
        const unsigned short* arow = W + (size_t)(m0 + n) * C;
        v8f acc = {};
#pragma unroll 4
        for (int kb = 0; kb < C; kb += 32) {
            __builtin_prefetch(arow + kb + 128, 0, 3);
            v16bf a = mk16(*(const u16x8*)(arow + kb + half * 8),
                           *(const u16x8*)(arow + kb + 16 + half * 8));
            v16bf bu = mk16(*(const u16x8*)(vin_b + kb + half * 8), // vector, all cols equal
                            *(const u16x8*)(vin_b + kb + 16 + half * 8));
            acc = __builtin_amdgcn_wmma_f32_16x16x32_bf16(false, a, false, bu,
                                                          (short)0, acc, false, false);
        }
        if (n == 0) {
#pragma unroll
            for (int r = 0; r < 8; ++r) {
                int m = m0 + half * 8 + r;
                float v = acc[r] + bias[m];
                if (res_f) v += res_f[m];
                if (leaky) v = v > 0.f ? v : 0.2f * v;
                if (out_b) out_b[m] = f2bf(v);
                if (out_f) out_f[m] = v;
            }
        }
    }
}

// ---------------- persistent rollout kernel ---------------------------------
__global__ void __launch_bounds__(TPB) seq_gen(
    const float* __restrict__ x, const float* __restrict__ conv_b,
    const float* __restrict__ res_b, const float* __restrict__ out_b,
    const unsigned short* __restrict__ Wc, const unsigned short* __restrict__ Wr,
    const unsigned short* __restrict__ Wo,
    unsigned short* bufA, unsigned short* bufB,   // body pointers (pad in front)
    unsigned short* zA, unsigned short* zB,       // body pointers (pad in front)
    float* h0f, unsigned short* h0b, float* h1f, unsigned short* h1b,
    unsigned short* h2b, float* out, const int* nsteps_p, unsigned* bar) {
    const int lane = threadIdx.x & 31;
    const int wave = (int)((blockIdx.x * blockDim.x + threadIdx.x) >> 5);
    const int gtid = (int)(blockIdx.x * blockDim.x + threadIdx.x);
    const int gstr = (int)(gridDim.x * blockDim.x);
    const int ns = *nsteps_p;

    for (int step = 0; step < ns; ++step) {
        unsigned short* bufP = (step & 1) ? bufB : bufA;
        unsigned short* bufN = (step & 1) ? bufA : bufB;

        conv_layer(Wc + 0 * MAT_ELEMS, Wc + 1 * MAT_ELEMS, conv_b + 0 * C, bufP, zA, 1, wave, lane);
        gsync(bar);
        conv_layer(Wc + 2 * MAT_ELEMS, Wc + 3 * MAT_ELEMS, conv_b + 1 * C, zA, zB, 2, wave, lane);
        gsync(bar);
        conv_layer(Wc + 4 * MAT_ELEMS, Wc + 5 * MAT_ELEMS, conv_b + 2 * C, zB, zA, 4, wave, lane);
        gsync(bar);
        conv_layer(Wc + 6 * MAT_ELEMS, Wc + 7 * MAT_ELEMS, conv_b + 3 * C, zA, zB, 8, wave, lane);
        gsync(bar);

        // head prep: h0 = x + z4[:, last]; also shift buffer rows 1..15 -> 0..14
        for (int c = gtid; c < C; c += gstr) {
            float v = x[c] + bf2f(zB[(size_t)(BUF - 1) * C + c]);
            h0f[c] = v;
            h0b[c] = f2bf(v);
        }
        for (int i = gtid; i < (BUF - 1) * C; i += gstr) bufN[i] = bufP[i + C];
        gsync(bar);

        matvec(Wr + 0 * MAT_ELEMS, res_b + 0 * C, h0b, h0f, h1b, h1f, 1, wave, lane);
        gsync(bar);
        matvec(Wr + 1 * MAT_ELEMS, res_b + 1 * C, h1b, h1f, h2b, nullptr, 1, wave, lane);
        gsync(bar);
        matvec(Wo, out_b, h2b, nullptr, bufN + (size_t)(BUF - 1) * C,
               out + (size_t)step * C, 0, wave, lane);
        gsync(bar);  // bufN row 15 + d_out visible before next step reads bufN
    }
}

// ---------------- weight conversion / init kernels --------------------------
__global__ void cvt_conv(const float* __restrict__ src, unsigned short* __restrict__ dst) {
    // dst[l][k][o][i]  <-  src[((l*C+o)*C+i)*2 + k]
    size_t total = 4ull * 2 * C * C;
    for (size_t dix = blockIdx.x * (size_t)blockDim.x + threadIdx.x; dix < total;
         dix += gridDim.x * (size_t)blockDim.x) {
        size_t i = dix & (C - 1);
        size_t o = (dix >> 11) & (C - 1);
        size_t k = (dix >> 22) & 1;
        size_t l = dix >> 23;
        dst[dix] = f2bf(src[(((l * C + o) * C + i) << 1) + k]);
    }
}
__global__ void cvt_mat(const float* __restrict__ src, unsigned short* __restrict__ dst,
                        size_t n) {
    for (size_t t = blockIdx.x * (size_t)blockDim.x + threadIdx.x; t < n;
         t += gridDim.x * (size_t)blockDim.x) dst[t] = f2bf(src[t]);
}
// zero: bufA fully (pad+body), pads of bufB/zA/zB, barrier. Base pointers here.
__global__ void initk(unsigned short* bufA0, unsigned short* bufB0,
                      unsigned short* zA0, unsigned short* zB0, unsigned* bar) {
    int tid = (int)(blockIdx.x * blockDim.x + threadIdx.x);
    int str = (int)(gridDim.x * blockDim.x);
    for (int i = tid; i < ZROWS * C; i += str) bufA0[i] = 0;
    for (int i = tid; i < ZPAD * C; i += str) { bufB0[i] = 0; zA0[i] = 0; zB0[i] = 0; }
    if (tid == 0) { bar[0] = 0u; bar[1] = 0u; }
}

// ---------------- launch -----------------------------------------------------
extern "C" void kernel_launch(void* const* d_in, const int* in_sizes, int n_in,
                              void* d_out, int out_size, void* d_ws, size_t ws_size,
                              hipStream_t stream) {
    const float* x      = (const float*)d_in[0];
    const float* conv_w = (const float*)d_in[1];
    const float* conv_b = (const float*)d_in[2];
    const float* res_w  = (const float*)d_in[3];
    const float* res_b  = (const float*)d_in[4];
    const float* out_w  = (const float*)d_in[5];
    const float* out_b  = (const float*)d_in[6];
    const int*   nsteps = (const int*)d_in[7];

    char* ws = (char*)d_ws;
    unsigned short* Wc    = (unsigned short*)ws; ws += 8 * MAT_ELEMS * sizeof(unsigned short);
    unsigned short* Wr    = (unsigned short*)ws; ws += 2 * MAT_ELEMS * sizeof(unsigned short);
    unsigned short* Wo    = (unsigned short*)ws; ws += 1 * MAT_ELEMS * sizeof(unsigned short);
    unsigned short* bufA0 = (unsigned short*)ws; ws += (size_t)ZROWS * C * sizeof(unsigned short);
    unsigned short* bufB0 = (unsigned short*)ws; ws += (size_t)ZROWS * C * sizeof(unsigned short);
    unsigned short* zA0   = (unsigned short*)ws; ws += (size_t)ZROWS * C * sizeof(unsigned short);
    unsigned short* zB0   = (unsigned short*)ws; ws += (size_t)ZROWS * C * sizeof(unsigned short);
    float*          h0f   = (float*)ws;          ws += (size_t)C * sizeof(float);
    unsigned short* h0b   = (unsigned short*)ws; ws += (size_t)C * sizeof(unsigned short);
    float*          h1f   = (float*)ws;          ws += (size_t)C * sizeof(float);
    unsigned short* h1b   = (unsigned short*)ws; ws += (size_t)C * sizeof(unsigned short);
    unsigned short* h2b   = (unsigned short*)ws; ws += (size_t)C * sizeof(unsigned short);
    unsigned*       bar   = (unsigned*)ws;       ws += 2 * sizeof(unsigned);

    cvt_conv<<<2048, 256, 0, stream>>>(conv_w, Wc);
    cvt_mat<<<1024, 256, 0, stream>>>(res_w, Wr, 2 * MAT_ELEMS);
    cvt_mat<<<512, 256, 0, stream>>>(out_w, Wo, MAT_ELEMS);
    initk<<<32, 256, 0, stream>>>(bufA0, bufB0, zA0, zB0, bar);

    seq_gen<<<NB, TPB, 0, stream>>>(x, conv_b, res_b, out_b, Wc, Wr, Wo,
                                    bufA0 + (size_t)ZPAD * C, bufB0 + (size_t)ZPAD * C,
                                    zA0 + (size_t)ZPAD * C, zB0 + (size_t)ZPAD * C,
                                    h0f, h0b, h1f, h1b, h2b,
                                    (float*)d_out, nsteps, bar);
}
